// GCN_Classifier_13640816132455
// MI455X (gfx1250) — compile-verified, matrix operand order checked by reading, not run on hardware
//
#include <hip/hip_runtime.h>
#include <math.h>

#define N_NODES 100000
#define N_EDGES 3200000
#define IN_DIM  128
#define HID_DIM 256
#define OUT_DIM 40
#define BN_EPS  1e-5f

typedef __attribute__((ext_vector_type(16))) _Float16 v16h;
typedef __attribute__((ext_vector_type(8)))  float    v8f;

// ---------------------------------------------------------------------------
// zero fill (accumulators must start at 0 each call; ws is poisoned 0xAA)
// ---------------------------------------------------------------------------
__global__ void zero_kernel(float* __restrict__ p, long n) {
    long i      = (long)blockIdx.x * blockDim.x + threadIdx.x;
    long stride = (long)gridDim.x * blockDim.x;
    for (; i < n; i += stride) p[i] = 0.0f;
}

// ---------------------------------------------------------------------------
// SpMM scatter-add: out[row[e], :] += vals[e] * x[col[e], :]
// One wave per edge; edge indices forced uniform (readfirstlane) so the
// row/col/val fetches become scalar (SMEM) loads. Feature gathers are
// L2-resident (feature matrix <= 102 MB < 192 MB L2). Scatter uses native
// global_atomic_add_f32 via unsafeAtomicAdd.
// ---------------------------------------------------------------------------
template<int F>
__global__ void spmm_kernel(const int*   __restrict__ row,
                            const int*   __restrict__ col,
                            const float* __restrict__ vals,
                            const float* __restrict__ x,
                            float*       __restrict__ out) {
    const int lane = threadIdx.x & 31;
    const int edge = blockIdx.x * (blockDim.x >> 5) + (threadIdx.x >> 5);
    if (edge >= N_EDGES) return;
    const int e = __builtin_amdgcn_readfirstlane(edge);  // wave-uniform

    const int   r = row[e];
    const int   c = col[e];
    const float v = vals[e];

    const float* xp = x   + (long)c * F;
    float*       op = out + (long)r * F;

#pragma unroll
    for (int i = 0; i < F / 128; ++i) {
        const int f = i * 128 + lane * 4;
        const float4 xv = *reinterpret_cast<const float4*>(xp + f);
        unsafeAtomicAdd(op + f + 0, v * xv.x);
        unsafeAtomicAdd(op + f + 1, v * xv.y);
        unsafeAtomicAdd(op + f + 2, v * xv.z);
        unsafeAtomicAdd(op + f + 3, v * xv.w);
    }
}

// ---------------------------------------------------------------------------
// Dense GEMM via V_WMMA_F32_16X16X32_F16, fused bias (+ BN + ReLU).
// out[M,N_OUT] = act( Z[M,K] @ W[K,N_OUT] + bias )
//
// Each wave computes FOUR 16x16 C tiles stacked in M, sharing one B fragment
// per K-step (4x register reuse of B). 8 waves / block -> 512 rows per block.
// No exec-divergent guards in the K loop: out-of-range rows/cols are index-
// clamped (their products only land in C elements that are never stored).
//
// Fragment layouts per CDNA5 ISA 7.12.2 (wave32):
//   A 16x32 f16: lane L -> row M = L&15 ; kbase = (L<16)?0:8 ;
//                halves h=0..7 -> K = kbase+h ; h=8..15 -> K = kbase+16+(h-8)
//   B 32x16 f16: lane L -> col N = L&15 ; same K mapping (symmetric)
//   C 16x16 f32: VGPR j : lanes 0-15 -> M=j ; lanes 16-31 -> M=j+8 ; N = L&15
// ---------------------------------------------------------------------------
template<int K_DIM, int N_OUT, bool BN_RELU>
__global__ void __launch_bounds__(256)
gemm_wmma_kernel(const float* __restrict__ Z,
                 const float* __restrict__ W,
                 const float* __restrict__ bias,
                 const float* __restrict__ gamma,
                 const float* __restrict__ beta,
                 const float* __restrict__ mean,
                 const float* __restrict__ var,
                 float*       __restrict__ out) {
    const int lane  = threadIdx.x & 31;
    const int wave  = threadIdx.x >> 5;
    const int tile0 = (blockIdx.x * 8 + wave) * 4;   // first of 4 M tiles
    if (tile0 * 16 >= N_NODES) return;               // wave-uniform exit
    const int col0  = blockIdx.y * 16;

    const int kb      = (lane & 16) ? 8 : 0;
    const int lane15  = lane & 15;
    const int bColRaw = col0 + lane15;
    const int bCol    = (bColRaw < N_OUT) ? bColRaw : (N_OUT - 1);  // clamp

    int aRow[4];
#pragma unroll
    for (int t = 0; t < 4; ++t) {
        const int rr = (tile0 + t) * 16 + lane15;
        aRow[t] = (rr < N_NODES) ? rr : (N_NODES - 1);              // clamp
    }

    v8f acc[4] = {{}, {}, {}, {}};

    for (int k0 = 0; k0 < K_DIM; k0 += 32) {
        // ---- B fragment: one column of W, shared by all 4 WMMAs ----
        v16h b;
        const float* wp = W + (long)(k0 + kb) * N_OUT + bCol;
#pragma unroll
        for (int h = 0; h < 8; ++h) {
            b[h]     = (_Float16)wp[(long)h * N_OUT];
            b[h + 8] = (_Float16)wp[(long)(h + 16) * N_OUT];
        }
        // ---- 4 A fragments (4x float4 contiguous loads each) + WMMA ----
#pragma unroll
        for (int t = 0; t < 4; ++t) {
            const float* zp = Z + (long)aRow[t] * K_DIM + k0 + kb;
            const float4 p0 = *reinterpret_cast<const float4*>(zp);
            const float4 p1 = *reinterpret_cast<const float4*>(zp + 4);
            const float4 p2 = *reinterpret_cast<const float4*>(zp + 16);
            const float4 p3 = *reinterpret_cast<const float4*>(zp + 20);
            v16h a;
            a[0]  = (_Float16)p0.x; a[1]  = (_Float16)p0.y;
            a[2]  = (_Float16)p0.z; a[3]  = (_Float16)p0.w;
            a[4]  = (_Float16)p1.x; a[5]  = (_Float16)p1.y;
            a[6]  = (_Float16)p1.z; a[7]  = (_Float16)p1.w;
            a[8]  = (_Float16)p2.x; a[9]  = (_Float16)p2.y;
            a[10] = (_Float16)p2.z; a[11] = (_Float16)p2.w;
            a[12] = (_Float16)p3.x; a[13] = (_Float16)p3.y;
            a[14] = (_Float16)p3.z; a[15] = (_Float16)p3.w;
            acc[t] = __builtin_amdgcn_wmma_f32_16x16x32_f16(
                         false, a, false, b, (short)0, acc[t], false, false);
        }
    }

    // ---- epilogue: bias (+ BN + ReLU), guarded store ----
    const int n = col0 + lane15;
    float bv = 0.0f, sc = 1.0f, sh = 0.0f;
    if (n < N_OUT) {
        bv = bias[n];
        if (BN_RELU) {
            const float inv = rsqrtf(var[n] + BN_EPS);
            sc = gamma[n] * inv;
            sh = beta[n] - mean[n] * sc;
        }
    }

#pragma unroll
    for (int t = 0; t < 4; ++t) {
        const int mBase = (tile0 + t) * 16 + ((lane & 16) ? 8 : 0);
#pragma unroll
        for (int j = 0; j < 8; ++j) {
            const int m = mBase + j;
            if (m < N_NODES && n < N_OUT) {
                float vv = acc[t][j] + bv;
                if (BN_RELU) {
                    vv = vv * sc + sh;
                    vv = fmaxf(vv, 0.0f);
                }
                out[(long)m * N_OUT + n] = vv;
            }
        }
    }
}

// ---------------------------------------------------------------------------
// log_softmax over 40 classes, one thread per row (rows are contiguous)
// ---------------------------------------------------------------------------
__global__ void logsoftmax_kernel(const float* __restrict__ logits,
                                  float*       __restrict__ out) {
    const int r = blockIdx.x * blockDim.x + threadIdx.x;
    if (r >= N_NODES) return;
    const float* p = logits + (long)r * OUT_DIM;
    float mx = -INFINITY;
#pragma unroll
    for (int i = 0; i < OUT_DIM; ++i) mx = fmaxf(mx, p[i]);
    float s = 0.0f;
#pragma unroll
    for (int i = 0; i < OUT_DIM; ++i) s += expf(p[i] - mx);
    const float lse = logf(s) + mx;
    float* o = out + (long)r * OUT_DIM;
#pragma unroll
    for (int i = 0; i < OUT_DIM; ++i) o[i] = p[i] - lse;
}

// ---------------------------------------------------------------------------
extern "C" void kernel_launch(void* const* d_in, const int* in_sizes, int n_in,
                              void* d_out, int out_size, void* d_ws, size_t ws_size,
                              hipStream_t stream) {
    const float* x      = (const float*)d_in[0];
    const int*   erow   = (const int*)  d_in[1];
    const int*   ecol   = (const int*)  d_in[2];
    const float* evals  = (const float*)d_in[3];
    const float* W1     = (const float*)d_in[4];
    const float* b1     = (const float*)d_in[5];
    const float* W2     = (const float*)d_in[6];
    const float* b2     = (const float*)d_in[7];
    const float* W3     = (const float*)d_in[8];
    const float* b3     = (const float*)d_in[9];
    const float* gamma1 = (const float*)d_in[10];
    const float* beta1  = (const float*)d_in[11];
    const float* mean1  = (const float*)d_in[12];
    const float* var1   = (const float*)d_in[13];
    const float* gamma2 = (const float*)d_in[14];
    const float* beta2  = (const float*)d_in[15];
    const float* mean2  = (const float*)d_in[16];
    const float* var2   = (const float*)d_in[17];

    float* bufA = (float*)d_ws;                         // 100000*256 f32 = 102.4 MB
    float* bufB = bufA + (size_t)N_NODES * HID_DIM;     // 102.4 MB

    const dim3 blk(256);
    const dim3 zgrid(2048);
    const int  spmmBlocks = (N_EDGES + 7) / 8;               // wave per edge
    const int  nTilesM    = (N_NODES + 15) / 16;             // 6250
    const int  mBlocks    = (nTilesM + 31) / 32;             // 196 (32 tiles/block)
    const dim3 gemmHidGrid(mBlocks, HID_DIM / 16);           // (196, 16)
    const dim3 gemmOutGrid(mBlocks, (OUT_DIM + 15) / 16);    // (196, 3)

    // ---- layer 1: z1 = A@x ; h1 = relu(bn(z1@W1+b1)) ----
    zero_kernel<<<zgrid, blk, 0, stream>>>(bufA, (long)N_NODES * IN_DIM);
    spmm_kernel<IN_DIM><<<spmmBlocks, blk, 0, stream>>>(erow, ecol, evals, x, bufA);
    gemm_wmma_kernel<IN_DIM, HID_DIM, true><<<gemmHidGrid, blk, 0, stream>>>(
        bufA, W1, b1, gamma1, beta1, mean1, var1, bufB);

    // ---- layer 2: z2 = A@h1 ; h2 = relu(bn(z2@W2+b2)) ----
    zero_kernel<<<zgrid, blk, 0, stream>>>(bufA, (long)N_NODES * HID_DIM);
    spmm_kernel<HID_DIM><<<spmmBlocks, blk, 0, stream>>>(erow, ecol, evals, bufB, bufA);
    gemm_wmma_kernel<HID_DIM, HID_DIM, true><<<gemmHidGrid, blk, 0, stream>>>(
        bufA, W2, b2, gamma2, beta2, mean2, var2, bufB);

    // ---- layer 3: z3 = (A@h2)@W3 + b3 ; out = log_softmax(z3) ----
    zero_kernel<<<zgrid, blk, 0, stream>>>(bufA, (long)N_NODES * HID_DIM);
    spmm_kernel<HID_DIM><<<spmmBlocks, blk, 0, stream>>>(erow, ecol, evals, bufB, bufA);
    gemm_wmma_kernel<HID_DIM, OUT_DIM, false><<<gemmOutGrid, blk, 0, stream>>>(
        bufA, W3, b3, nullptr, nullptr, nullptr, nullptr, bufB);

    logsoftmax_kernel<<<(N_NODES + 255) / 256, blk, 0, stream>>>(bufB, (float*)d_out);
}
